// Model_87660282511615
// MI455X (gfx1250) — matrix operand order repaired and checked by reading, not compiled, not run on hardware
//
#include <hip/hip_runtime.h>
#include <stdint.h>

// ---------------------------------------------------------------------------
// e3nn-style 5-layer equivariant graph conv for MI455X (gfx1250, wave32).
// Memory/atomic bound => fuse per-layer into: edge kernel (gather + tensor
// product + WMMA channel-mixing + f32 atomic scatter) and node kernel
// (self-connection + gate).  Per-edge GEMMs run as v_wmma_f32_16x16x32_f16
// (f16 A staged in LDS, f16 B built in registers, f32 accumulate) on 16-edge tiles, one wave/tile.
// Normalization constants are pre-folded into the f16 weight fragments.
// Scatter addresses are built as (dst*64 | chan) -- bit-disjoint because the
// accumulator stride is 64 and chan < 64 -- so every atomic lowers to a single
// v_or_b32 + SADDR-form global_atomic_add_f32 with scale_offset.
// ---------------------------------------------------------------------------

typedef __attribute__((ext_vector_type(16))) _Float16 v16h;
typedef __attribute__((ext_vector_type(8)))  float    v8f;

#define FSTR 52          // node feature stride (floats): 0e[10] 0o[10] 1e[5][3] 1o[5][3]
#define ASTR 64          // conv accumulator stride (floats); power of two for OR-addressing
#define A0E 0
#define A0O 20
#define A1E 30
#define A1O 45
#define F0E 0
#define F0O 10
#define F1E 20
#define F1O 35

#define ISQ3   0.57735026918962576f
#define ISQ2   0.70710678118654752f
#define INVNN  0.81649658092772603f   // 1/sqrt(1.5)
#define S_SC15 0.21081851067789195f   // INVNN / sqrt(15)
#define S_V20  0.18257418583505536f   // INVNN / sqrt(20)
#define RS10   0.31622776601683794f   // 1/sqrt(10)
#define RS5    0.44721359549995794f   // 1/sqrt(5)

union Frag16 { v16h v; uint32_t u[8]; };
union Pack2  { uint32_t u; _Float16 h[2]; };

__device__ __forceinline__ void atomAddF(float* base, uint32_t off, float v) {
  __hip_atomic_fetch_add(base + off, v, __ATOMIC_RELAXED, __HIP_MEMORY_SCOPE_AGENT);
}

__device__ __forceinline__ float geluf(float x) {
  const float k0 = 0.7978845608028654f;  // sqrt(2/pi)
  return 0.5f * x * (1.0f + tanhf(k0 * (x + 0.044715f * x * x * x)));
}
__device__ __forceinline__ float sigmoidf(float x) { return 1.0f / (1.0f + expf(-x)); }

__device__ __forceinline__ v8f wmma_f32_f16(v16h a, v16h b, v8f c) {
  return __builtin_amdgcn_wmma_f32_16x16x32_f16(false, a, false, b, (short)0, c, false, false);
}

// B fragment (K x Nout weights, row-major f32 in global, converted to f16 with
// the path-normalization constant pre-folded in):
// VGPR j, lanes 0-15 hold K=2j,2j+1 ; lanes 16-31 hold K=16+2j,17+2j (N = lane%16).
__device__ __forceinline__ v16h load_b_frag(const float* __restrict__ W, int M, int Nout,
                                            int colbase, float scale, int lane) {
  Frag16 b;
  const int half = lane >> 4;
  const int n = colbase + (lane & 15);
#pragma unroll
  for (int j = 0; j < 8; ++j) {
    const int m0 = 2 * j + 16 * half;
    Pack2 p;
    p.h[0] = (m0     < M && n < Nout) ? (_Float16)(W[m0 * Nout + n] * scale)       : (_Float16)0.f;
    p.h[1] = (m0 + 1 < M && n < Nout) ? (_Float16)(W[(m0 + 1) * Nout + n] * scale) : (_Float16)0.f;
    b.u[j] = p.u;
  }
  return b.v;
}

// A fragment from LDS row (one edge, one channel group, K padded/zeroed to 32).
// A k order used here (contiguous, matches B): VGPR j, lanes 0-15 -> K=2j,2j+1 ;
// lanes 16-31: same +16.
__device__ __forceinline__ v16h load_a_frag(const _Float16* __restrict__ p, int half) {
  Frag16 a;
#pragma unroll
  for (int j = 0; j < 8; ++j) {
    const int kb = 2 * j + 16 * half;
    a.u[j] = *(const uint32_t*)(p + kb);
  }
  return a.v;
}

// ---------------------------------------------------------------------------
// Layer 0 edge conv: input 0e x1.  Scalar (work per edge is tiny; atomics dominate).
// ---------------------------------------------------------------------------
__global__ void edge_l0_kernel(const float* __restrict__ x, const int* __restrict__ esrc,
                               const int* __restrict__ edst, const float* __restrict__ attr,
                               const float* __restrict__ Wtp0e, const float* __restrict__ Wtp1o,
                               float* __restrict__ acc, int E) {
  int ei = blockIdx.x * blockDim.x + threadIdx.x;
  if (ei >= E) return;
  float4 at = ((const float4*)attr)[ei];
  float xs = x[esrc[ei]];
  const uint32_t ob = (uint32_t)edst[ei] << 6;       // * ASTR
  float c0 = xs * at.x * INVNN;                      // nrm = 1/sqrt(1) = 1
#pragma unroll
  for (int k = 0; k < 20; ++k) atomAddF(acc, ob | (uint32_t)(A0E + k), c0 * Wtp0e[k]);
  float cx = xs * at.y * INVNN, cy = xs * at.z * INVNN, cz = xs * at.w * INVNN;
#pragma unroll
  for (int k = 0; k < 5; ++k) {
    float w = Wtp1o[k];
    atomAddF(acc, ob | (uint32_t)(A1O + 3 * k + 0), cx * w);
    atomAddF(acc, ob | (uint32_t)(A1O + 3 * k + 1), cy * w);
    atomAddF(acc, ob | (uint32_t)(A1O + 3 * k + 2), cz * w);
  }
}

// ---------------------------------------------------------------------------
// Mid layers (1..3) edge conv: one wave per 16-edge tile, WMMA channel mixing.
// TP channels (K): 0e:15, 0o:15, 1e:20 x3 comps, 1o:20 x3 comps -> 8 groups.
// ---------------------------------------------------------------------------
#define EMS 290   // per-edge LDS stride in halves (8 groups * 36 + pad; bank-friendly)

__global__ __launch_bounds__(32) void edge_mid_kernel(
    const float* __restrict__ fbuf, const int* __restrict__ esrc,
    const int* __restrict__ edst, const float* __restrict__ attr,
    const float* __restrict__ Wtp0e, const float* __restrict__ Wtp0o,
    const float* __restrict__ Wtp1e, const float* __restrict__ Wtp1o,
    float* __restrict__ acc, int ntiles) {
  __shared__ _Float16 sch[16 * EMS];
  __shared__ int sdst[16];
  const int lane = threadIdx.x;
  const int e = lane & 15;
  const int half = lane >> 4;
  const int tile = blockIdx.x;
  if (tile >= ntiles) return;

  // Weight fragments (constant per kernel): 5 B-fragments, norm folded in.
  const v16h b0e0 = load_b_frag(Wtp0e, 15, 20, 0,  S_SC15, lane);
  const v16h b0e1 = load_b_frag(Wtp0e, 15, 20, 16, S_SC15, lane);
  const v16h b0o  = load_b_frag(Wtp0o, 15, 10, 0,  S_SC15, lane);
  const v16h b1e  = load_b_frag(Wtp1e, 20, 5, 0,   S_V20,  lane);
  const v16h b1o  = load_b_frag(Wtp1o, 20, 5, 0,   S_V20,  lane);

  const int base = tile * 16;
  const float4 at = ((const float4*)attr)[base + e];
  const float s = at.x, vx = at.y, vy = at.z, vz = at.w;
  const int src = esrc[base + e];
  if (half == 0) sdst[e] = edst[base + e];

  // Zero-fill K in [0,32) for all 8 groups x 16 edges (2048 dwords).
#pragma unroll
  for (int t = 0; t < 64; ++t) {
    const int idx = t * 32 + lane;
    const int k2 = idx & 15, grp = (idx >> 4) & 7, ee = idx >> 7;
    *(uint32_t*)&sch[ee * EMS + grp * 36 + 2 * k2] = 0u;
  }

  const float* __restrict__ f = fbuf + (size_t)src * FSTR;
  _Float16* __restrict__ row = sch + e * EMS;
  if (half == 0) {
    // channels [0..9] of every group come from the scalar inputs
#pragma unroll
    for (int m = 0; m < 10; ++m) {
      float f0e = f[F0E + m], f0o = f[F0O + m];
      row[0 * 36 + m] = (_Float16)(f0e * s);
      row[1 * 36 + m] = (_Float16)(f0o * s);
      row[2 * 36 + m] = (_Float16)(f0o * vx);
      row[3 * 36 + m] = (_Float16)(f0o * vy);
      row[4 * 36 + m] = (_Float16)(f0o * vz);
      row[5 * 36 + m] = (_Float16)(f0e * vx);
      row[6 * 36 + m] = (_Float16)(f0e * vy);
      row[7 * 36 + m] = (_Float16)(f0e * vz);
    }
  } else {
    // channels [10..] come from the vector inputs (dot / *s / cross terms)
#pragma unroll
    for (int m = 0; m < 5; ++m) {
      float ex = f[F1E + 3 * m], ey = f[F1E + 3 * m + 1], ez = f[F1E + 3 * m + 2];
      float ox = f[F1O + 3 * m], oy = f[F1O + 3 * m + 1], oz = f[F1O + 3 * m + 2];
      row[0 * 36 + 10 + m] = (_Float16)((ox * vx + oy * vy + oz * vz) * ISQ3);
      row[1 * 36 + 10 + m] = (_Float16)((ex * vx + ey * vy + ez * vz) * ISQ3);
      row[2 * 36 + 10 + m] = (_Float16)(ex * s);
      row[3 * 36 + 10 + m] = (_Float16)(ey * s);
      row[4 * 36 + 10 + m] = (_Float16)(ez * s);
      row[2 * 36 + 15 + m] = (_Float16)((oy * vz - oz * vy) * ISQ2);
      row[3 * 36 + 15 + m] = (_Float16)((oz * vx - ox * vz) * ISQ2);
      row[4 * 36 + 15 + m] = (_Float16)((ox * vy - oy * vx) * ISQ2);
      row[5 * 36 + 10 + m] = (_Float16)((ey * vz - ez * vy) * ISQ2);
      row[6 * 36 + 10 + m] = (_Float16)((ez * vx - ex * vz) * ISQ2);
      row[7 * 36 + 10 + m] = (_Float16)((ex * vy - ey * vx) * ISQ2);
      row[5 * 36 + 15 + m] = (_Float16)(ox * s);
      row[6 * 36 + 15 + m] = (_Float16)(oy * s);
      row[7 * 36 + 15 + m] = (_Float16)(oz * s);
    }
  }
  __syncthreads();

  // Per-row destination base offsets (C/D rows: M=r for lanes 0-15, M=r+8 above).
  uint32_t ofs[8];
#pragma unroll
  for (int r = 0; r < 8; ++r) ofs[r] = (uint32_t)sdst[r + 8 * half] << 6;  // * ASTR

  const v8f zc = {};
  const uint32_t c = (uint32_t)(lane & 15);  // output channel within the D tile
  v16h a;
  v8f d;

  // 0e: (16x32) x (32x20) -> two N tiles
  a = load_a_frag(sch + e * EMS + 0 * 36, half);
  d = wmma_f32_f16(a, b0e0, zc);
#pragma unroll
  for (int r = 0; r < 8; ++r) atomAddF(acc, ofs[r] | (A0E + c), d[r]);
  d = wmma_f32_f16(a, b0e1, zc);
  if (c < 4) {
#pragma unroll
    for (int r = 0; r < 8; ++r) atomAddF(acc, ofs[r] | (A0E + 16 + c), d[r]);
  }
  // 0o
  a = load_a_frag(sch + e * EMS + 1 * 36, half);
  d = wmma_f32_f16(a, b0o, zc);
  if (c < 10) {
#pragma unroll
    for (int r = 0; r < 8; ++r) atomAddF(acc, ofs[r] | (A0O + c), d[r]);
  }
  // 1e / 1o: 3 components each share one B fragment
#pragma unroll
  for (int i = 0; i < 3; ++i) {
    a = load_a_frag(sch + e * EMS + (2 + i) * 36, half);
    d = wmma_f32_f16(a, b1e, zc);
    if (c < 5) {
      const uint32_t co = A1E + 3 * c + (uint32_t)i;   // <= 44 < 64
#pragma unroll
      for (int r = 0; r < 8; ++r) atomAddF(acc, ofs[r] | co, d[r]);
    }
  }
#pragma unroll
  for (int i = 0; i < 3; ++i) {
    a = load_a_frag(sch + e * EMS + (5 + i) * 36, half);
    d = wmma_f32_f16(a, b1o, zc);
    if (c < 5) {
      const uint32_t co = A1O + 3 * c + (uint32_t)i;   // <= 59 < 64
#pragma unroll
      for (int r = 0; r < 8; ++r) atomAddF(acc, ofs[r] | co, d[r]);
    }
  }
}

// ---------------------------------------------------------------------------
// Last layer (4) edge conv: scalars only (0e:6, 0o:1), WMMA.
// acc layout for this layer: [0]=0o, [1..6]=0e  (matches output order).
// ---------------------------------------------------------------------------
#define ELS 74

__global__ __launch_bounds__(32) void edge_last_kernel(
    const float* __restrict__ fbuf, const int* __restrict__ esrc,
    const int* __restrict__ edst, const float* __restrict__ attr,
    const float* __restrict__ Wtp0e, const float* __restrict__ Wtp0o,
    float* __restrict__ acc, int ntiles) {
  __shared__ _Float16 sch[16 * ELS];
  __shared__ int sdst[16];
  const int lane = threadIdx.x;
  const int e = lane & 15;
  const int half = lane >> 4;
  const int tile = blockIdx.x;
  if (tile >= ntiles) return;

  const v16h b0e = load_b_frag(Wtp0e, 15, 6, 0, S_SC15, lane);
  const v16h b0o = load_b_frag(Wtp0o, 15, 1, 0, S_SC15, lane);

  const int base = tile * 16;
  const float4 at = ((const float4*)attr)[base + e];
  const float s = at.x, vx = at.y, vy = at.z, vz = at.w;
  const int src = esrc[base + e];
  if (half == 0) sdst[e] = edst[base + e];

#pragma unroll
  for (int t = 0; t < 16; ++t) {
    const int idx = t * 32 + lane;   // 512 dwords: 16 edges * 2 groups * 16
    const int k2 = idx & 15, grp = (idx >> 4) & 1, ee = idx >> 5;
    *(uint32_t*)&sch[ee * ELS + grp * 36 + 2 * k2] = 0u;
  }

  const float* __restrict__ f = fbuf + (size_t)src * FSTR;
  _Float16* __restrict__ row = sch + e * ELS;
  if (half == 0) {
#pragma unroll
    for (int m = 0; m < 10; ++m) {
      row[0 * 36 + m] = (_Float16)(f[F0E + m] * s);
      row[1 * 36 + m] = (_Float16)(f[F0O + m] * s);
    }
  } else {
#pragma unroll
    for (int m = 0; m < 5; ++m) {
      float ex = f[F1E + 3 * m], ey = f[F1E + 3 * m + 1], ez = f[F1E + 3 * m + 2];
      float ox = f[F1O + 3 * m], oy = f[F1O + 3 * m + 1], oz = f[F1O + 3 * m + 2];
      row[0 * 36 + 10 + m] = (_Float16)((ox * vx + oy * vy + oz * vz) * ISQ3);
      row[1 * 36 + 10 + m] = (_Float16)((ex * vx + ey * vy + ez * vz) * ISQ3);
    }
  }
  __syncthreads();

  uint32_t ofs[8];
#pragma unroll
  for (int r = 0; r < 8; ++r) ofs[r] = (uint32_t)sdst[r + 8 * half] << 6;

  const v8f zc = {};
  const uint32_t c = (uint32_t)(lane & 15);
  v16h a;
  v8f d;

  a = load_a_frag(sch + e * ELS + 0 * 36, half);
  d = wmma_f32_f16(a, b0e, zc);
  if (c < 6) {
#pragma unroll
    for (int r = 0; r < 8; ++r) atomAddF(acc, ofs[r] | (1 + c), d[r]);
  }
  a = load_a_frag(sch + e * ELS + 1 * 36, half);
  d = wmma_f32_f16(a, b0o, zc);
  if (c == 0) {
#pragma unroll
    for (int r = 0; r < 8; ++r) atomAddF(acc, ofs[r], d[r]);
  }
}

// ---------------------------------------------------------------------------
// Node kernels: self-connection + gate.
// ---------------------------------------------------------------------------
__global__ void node_l0_kernel(const float* __restrict__ x, const float* __restrict__ acc,
                               const float* __restrict__ Wsc0e, float* __restrict__ fbuf, int N) {
  int n = blockIdx.x * blockDim.x + threadIdx.x;
  if (n >= N) return;
  const float* a = acc + (size_t)n * ASTR;
  float xv = x[n];
  float* fo = fbuf + (size_t)n * FSTR;
  float g[10];
#pragma unroll
  for (int m = 0; m < 10; ++m) fo[F0E + m] = geluf(a[A0E + m] + xv * Wsc0e[m]);
#pragma unroll
  for (int m = 0; m < 10; ++m) g[m] = sigmoidf(a[A0E + 10 + m] + xv * Wsc0e[10 + m]);
#pragma unroll
  for (int m = 0; m < 10; ++m) fo[F0O + m] = 0.f;     // tanh(0) = 0
#pragma unroll
  for (int k = 0; k < 15; ++k) fo[F1E + k] = 0.f;     // zero 1e * gate
#pragma unroll
  for (int m = 0; m < 5; ++m)
#pragma unroll
    for (int i = 0; i < 3; ++i) fo[F1O + 3 * m + i] = a[A1O + 3 * m + i] * g[5 + m];
}

__global__ void node_mid_kernel(float* __restrict__ fbuf, const float* __restrict__ acc,
                                const float* __restrict__ Wsc0e, const float* __restrict__ Wsc0o,
                                const float* __restrict__ Wsc1e, const float* __restrict__ Wsc1o,
                                int N) {
  int n = blockIdx.x * blockDim.x + threadIdx.x;
  if (n >= N) return;
  float* fp = fbuf + (size_t)n * FSTR;
  float f0e[10], f0o[10], f1e[15], f1o[15];
#pragma unroll
  for (int m = 0; m < 10; ++m) { f0e[m] = fp[F0E + m]; f0o[m] = fp[F0O + m]; }
#pragma unroll
  for (int k = 0; k < 15; ++k) { f1e[k] = fp[F1E + k]; f1o[k] = fp[F1O + k]; }
  const float* a = acc + (size_t)n * ASTR;

  float c0e[20];
#pragma unroll
  for (int k = 0; k < 20; ++k) {
    float t = 0.f;
#pragma unroll
    for (int m = 0; m < 10; ++m) t += f0e[m] * Wsc0e[m * 20 + k];
    c0e[k] = a[A0E + k] + t * RS10;
  }
  float g[10];
#pragma unroll
  for (int m = 0; m < 10; ++m) fp[F0E + m] = geluf(c0e[m]);
#pragma unroll
  for (int m = 0; m < 10; ++m) g[m] = sigmoidf(c0e[10 + m]);
#pragma unroll
  for (int k = 0; k < 10; ++k) {
    float t = 0.f;
#pragma unroll
    for (int m = 0; m < 10; ++m) t += f0o[m] * Wsc0o[m * 10 + k];
    fp[F0O + k] = tanhf(a[A0O + k] + t * RS10);
  }
#pragma unroll
  for (int k = 0; k < 5; ++k)
#pragma unroll
    for (int i = 0; i < 3; ++i) {
      float t = 0.f;
#pragma unroll
      for (int m = 0; m < 5; ++m) t += f1e[3 * m + i] * Wsc1e[m * 5 + k];
      fp[F1E + 3 * k + i] = (a[A1E + 3 * k + i] + t * RS5) * g[k];
    }
#pragma unroll
  for (int k = 0; k < 5; ++k)
#pragma unroll
    for (int i = 0; i < 3; ++i) {
      float t = 0.f;
#pragma unroll
      for (int m = 0; m < 5; ++m) t += f1o[3 * m + i] * Wsc1o[m * 5 + k];
      fp[F1O + 3 * k + i] = (a[A1O + 3 * k + i] + t * RS5) * g[5 + k];
    }
}

__global__ void node_final_kernel(const float* __restrict__ fbuf, const float* __restrict__ acc,
                                  const float* __restrict__ Wsc0e, const float* __restrict__ Wsc0o,
                                  float* __restrict__ out, int N) {
  int n = blockIdx.x * blockDim.x + threadIdx.x;
  if (n >= N) return;
  const float* fp = fbuf + (size_t)n * FSTR;
  const float* a = acc + (size_t)n * ASTR;
  float t = 0.f;
#pragma unroll
  for (int m = 0; m < 10; ++m) t += fp[F0O + m] * Wsc0o[m];
  out[n * 7 + 0] = a[0] + t * RS10;
#pragma unroll
  for (int k = 0; k < 6; ++k) {
    float u = 0.f;
#pragma unroll
    for (int m = 0; m < 10; ++m) u += fp[F0E + m] * Wsc0e[m * 6 + k];
    out[n * 7 + 1 + k] = a[1 + k] + u * RS10;
  }
}

// ---------------------------------------------------------------------------
// Host-side orchestration (all on `stream`, graph-capture safe).
// d_in order (setup_inputs dict order, params flattened by sorted pytree keys):
//  0:x  1:edge_src  2:edge_dst  3:edge_attr
//  4..6   : L0  sc0e(1x20) | tp0e(1x20) tp1o(1x5)
//  7..30  : L1..L3 each: sc0e(10x20) sc0o(10x10) sc1e(5x5) sc1o(5x5)
//                         tp0e(15x20) tp0o(15x10) tp1e(20x5) tp1o(20x5)
//  31..34 : L4  sc0e(10x6) sc0o(10x1) | tp0e(15x6) tp0o(15x1)
// ---------------------------------------------------------------------------
extern "C" void kernel_launch(void* const* d_in, const int* in_sizes, int n_in,
                              void* d_out, int out_size, void* d_ws, size_t ws_size,
                              hipStream_t stream) {
  const float* x    = (const float*)d_in[0];
  const int*   esrc = (const int*)d_in[1];
  const int*   edst = (const int*)d_in[2];
  const float* attr = (const float*)d_in[3];
  const int N = in_sizes[0];
  const int E = in_sizes[1];
  const int ntiles = E / 16;

  float* acc  = (float*)d_ws;
  float* fbuf = (float*)((char*)d_ws + (size_t)N * ASTR * sizeof(float));

  const int NB = (N + 255) / 256;

  // ---- layer 0 ----
  hipMemsetAsync(acc, 0, (size_t)N * ASTR * sizeof(float), stream);
  edge_l0_kernel<<<(E + 255) / 256, 256, 0, stream>>>(
      x, esrc, edst, attr, (const float*)d_in[5], (const float*)d_in[6], acc, E);
  node_l0_kernel<<<NB, 256, 0, stream>>>(x, acc, (const float*)d_in[4], fbuf, N);

  // ---- layers 1..3 ----
  for (int L = 1; L <= 3; ++L) {
    const int b = 7 + (L - 1) * 8;
    const float* sc0e = (const float*)d_in[b + 0];
    const float* sc0o = (const float*)d_in[b + 1];
    const float* sc1e = (const float*)d_in[b + 2];
    const float* sc1o = (const float*)d_in[b + 3];
    const float* tp0e = (const float*)d_in[b + 4];
    const float* tp0o = (const float*)d_in[b + 5];
    const float* tp1e = (const float*)d_in[b + 6];
    const float* tp1o = (const float*)d_in[b + 7];
    hipMemsetAsync(acc, 0, (size_t)N * ASTR * sizeof(float), stream);
    edge_mid_kernel<<<ntiles, 32, 0, stream>>>(fbuf, esrc, edst, attr,
                                               tp0e, tp0o, tp1e, tp1o, acc, ntiles);
    node_mid_kernel<<<NB, 256, 0, stream>>>(fbuf, acc, sc0e, sc0o, sc1e, sc1o, N);
  }

  // ---- layer 4 ----
  hipMemsetAsync(acc, 0, (size_t)N * ASTR * sizeof(float), stream);
  edge_last_kernel<<<ntiles, 32, 0, stream>>>(fbuf, esrc, edst, attr,
                                              (const float*)d_in[33],
                                              (const float*)d_in[34], acc, ntiles);
  node_final_kernel<<<NB, 256, 0, stream>>>(fbuf, acc,
                                            (const float*)d_in[31],
                                            (const float*)d_in[32],
                                            (float*)d_out, N);
}
